// GNNStack_38749194945020
// MI455X (gfx1250) — matrix operand order repaired and checked by reading, not compiled, not run on hardware
//
#include <hip/hip_runtime.h>
#include <hip/hip_bf16.h>

typedef __attribute__((ext_vector_type(16))) _Float16 v16h;
typedef __attribute__((ext_vector_type(8)))  float    v8f;

#define N_NODES 20000
#define E_EDGES 500000
#define D_IN    128
#define HC      192

__device__ __forceinline__ unsigned orderF(float f) {
    unsigned u = __float_as_uint(f);
    return (u & 0x80000000u) ? ~u : (u | 0x80000000u);
}
__device__ __forceinline__ float unorderF(unsigned u) {
    unsigned v = (u & 0x80000000u) ? (u ^ 0x80000000u) : ~u;
    return __uint_as_float(v);
}
__device__ __forceinline__ float lrelu(float x, float s) { return x > 0.f ? x : s * x; }

// -------- GEMM: C[M,N] = A[M,K] @ B[K,N], f32 via split-f16 WMMA (3x per tile) ----
// lda = K, ldb = ldc = N. M%16==0, N%16==0, K%32==0. One wave per 16x16 C tile.
__global__ __launch_bounds__(128) void gemm_wmma_f32(
    const float* __restrict__ A, const float* __restrict__ B, float* __restrict__ C,
    int M, int N, int K)
{
    int wave = blockIdx.x * (blockDim.x >> 5) + (threadIdx.x >> 5);
    int lane = threadIdx.x & 31;
    int nTiles = N >> 4;
    int mt = wave / nTiles, nt = wave % nTiles;
    int row0 = mt << 4, col0 = nt << 4;
    int lm = lane & 15;     // sub-index within 16
    int g  = lane >> 4;     // lane group 0/1
    v8f acc = {};
    const float* arow = A + (size_t)(row0 + lm) * K;
    for (int k0 = 0; k0 < K; k0 += 32) {
        v16h ah, al, bh, bl;
#pragma unroll
        for (int j = 0; j < 16; ++j) {
            // A 16x32 f16 layout: lane grp g, elem j -> K = j + g*16
            int ka = k0 + g * 16 + j;
            float av = arow[ka];
            _Float16 avh = (_Float16)av;
            ah[j] = avh;
            al[j] = (_Float16)(av - (float)avh);
            // B 32x16 f16 layout: lane grp g, elem j -> K = j + g*16, N = lm
            int kb = k0 + (g << 4) + j;
            float bv = B[(size_t)kb * N + col0 + lm];
            _Float16 bvh = (_Float16)bv;
            bh[j] = bvh;
            bl[j] = (_Float16)(bv - (float)bvh);
        }
        acc = __builtin_amdgcn_wmma_f32_16x16x32_f16(false, ah, false, bh, (short)0, acc, false, false);
        acc = __builtin_amdgcn_wmma_f32_16x16x32_f16(false, ah, false, bl, (short)0, acc, false, false);
        acc = __builtin_amdgcn_wmma_f32_16x16x32_f16(false, al, false, bh, (short)0, acc, false, false);
    }
#pragma unroll
    for (int r = 0; r < 8; ++r)   // C/D: lane L, vgpr r -> M = r + 8*(L/16), N = L%16
        C[(size_t)(row0 + r + g * 8) * N + col0 + lm] = acc[r];
}

// -------- out[n,h] = sum_c H[n, h*64+c] * avec[h*64+c] --------------------------
__global__ void rowdot_head(const float* __restrict__ Hm, const float* __restrict__ avec,
                            float* __restrict__ out, int Nn)
{
    int n = blockIdx.x * blockDim.x + threadIdx.x;
    if (n >= Nn) return;
    const float* r = Hm + (size_t)n * HC;
#pragma unroll
    for (int h = 0; h < 3; ++h) {
        float a = 0.f;
        for (int c = 0; c < 64; ++c) a += r[h * 64 + c] * avec[h * 64 + c];
        out[n * 3 + h] = a;
    }
}

// -------- out[n,h] = sum_k X[n,k] * U[k*3+h] ------------------------------------
__global__ void matvec3(const float* __restrict__ X, const float* __restrict__ U,
                        float* __restrict__ out, int Nn)
{
    int n = blockIdx.x * blockDim.x + threadIdx.x;
    if (n >= Nn) return;
    const float* r = X + (size_t)n * HC;
    float a0 = 0.f, a1 = 0.f, a2 = 0.f;
    for (int k = 0; k < HC; ++k) {
        float x = r[k];
        a0 += x * U[k * 3 + 0]; a1 += x * U[k * 3 + 1]; a2 += x * U[k * 3 + 2];
    }
    out[n * 3 + 0] = a0; out[n * 3 + 1] = a1; out[n * 3 + 2] = a2;
}

// -------- precompute U2[k,h] = sum_c Wd2[k,h*64+c]*ad2[h,c]; ce[0..2]=layer1, [3..5]=layer2
__global__ void prep_consts(const float* __restrict__ Wd2, const float* __restrict__ ad2,
                            const float* __restrict__ We1, const float* __restrict__ ae1,
                            const float* __restrict__ We2, const float* __restrict__ ae2,
                            float* __restrict__ U, float* __restrict__ ce)
{
    int t = threadIdx.x;  // 0..191
#pragma unroll
    for (int h = 0; h < 3; ++h) {
        float a = 0.f;
        for (int c = 0; c < 64; ++c) a += Wd2[(size_t)t * HC + h * 64 + c] * ad2[h * 64 + c];
        U[t * 3 + h] = a;
    }
    if (t < 3) {
        float a = 0.f, b = 0.f;
        for (int c = 0; c < 64; ++c) {
            a += We1[t * 64 + c] * ae1[t * 64 + c];
            b += We2[t * 64 + c] * ae2[t * 64 + c];
        }
        ce[t] = a; ce[3 + t] = b;
    }
}

// -------- init accumulators / S / ordered-max buffers ---------------------------
__global__ void init_bufs(float* a0, float* a1, float* S0, float* S1,
                          unsigned* M0, unsigned* M1, int nAcc, int nNode)
{
    unsigned ninf = orderF(-__builtin_inff());
    for (int i = blockIdx.x * blockDim.x + threadIdx.x; i < nAcc; i += gridDim.x * blockDim.x) {
        a0[i] = 0.f; a1[i] = 0.f;
        if (i < nNode) { S0[i] = 0.f; S1[i] = 0.f; M0[i] = ninf; M1[i] = ninf; }
    }
}

// -------- edge pass A: segment max of leaky(alpha) via ordered-uint atomicMax ---
__global__ void edge_max(const int* __restrict__ src, const int* __restrict__ dst,
                         const float* __restrict__ asrc, const float* __restrict__ adst,
                         const float* __restrict__ ea, const float* __restrict__ ce,
                         unsigned* __restrict__ Mb, int Ne)
{
    int e = blockIdx.x * blockDim.x + threadIdx.x;
    if (e >= Ne) return;
    int s = src[e], d = dst[e];
    float w = ea[e];
#pragma unroll
    for (int h = 0; h < 3; ++h) {
        float a = lrelu(asrc[s * 3 + h] + adst[d * 3 + h] + w * ce[h], 0.2f);
        atomicMax(&Mb[d * 3 + h], orderF(a));
    }
}

// -------- edge pass B: e=exp(a-m), S += e, acc[dst] += e * Hs[src] (192 wide) ---
__global__ __launch_bounds__(192) void edge_acc(
    const int* __restrict__ src, const int* __restrict__ dst,
    const float* __restrict__ asrc, const float* __restrict__ adst,
    const float* __restrict__ ea, const float* __restrict__ ce,
    const unsigned* __restrict__ Mb, float* __restrict__ Sb,
    const float* __restrict__ Hs, float* __restrict__ acc)
{
    __shared__ float eh[3];
    int e = blockIdx.x;
    int s = src[e], d = dst[e];
    int t = threadIdx.x;
    if (t < 3) {
        float a = lrelu(asrc[s * 3 + t] + adst[d * 3 + t] + ea[e] * ce[t], 0.2f);
        float m = unorderF(Mb[d * 3 + t]);
        float ev = __expf(a - m);
        atomicAdd(&Sb[d * 3 + t], ev);
        eh[t] = ev;
    }
    __syncthreads();
    float ev = eh[t >> 6];
    atomicAdd(&acc[(size_t)d * HC + t], ev * Hs[(size_t)s * HC + t]);
}

// -------- edge pass D: write normalized attention -------------------------------
__global__ void edge_alpha(const int* __restrict__ src, const int* __restrict__ dst,
                           const float* __restrict__ asrc, const float* __restrict__ adst,
                           const float* __restrict__ ea, const float* __restrict__ ce,
                           const unsigned* __restrict__ Mb, const float* __restrict__ Sb,
                           float* __restrict__ at, int Ne)
{
    int e = blockIdx.x * blockDim.x + threadIdx.x;
    if (e >= Ne) return;
    int s = src[e], d = dst[e];
    float w = ea[e];
#pragma unroll
    for (int h = 0; h < 3; ++h) {
        float a = lrelu(asrc[s * 3 + h] + adst[d * 3 + h] + w * ce[h], 0.2f);
        float ev = __expf(a - unorderF(Mb[d * 3 + h]));
        at[(size_t)e * 3 + h] = ev / (Sb[d * 3 + h] + 1e-16f);
    }
}

// -------- layer1 node pass: normalize, +b, leaky 0.01, LayerNorm(192) in place --
__global__ __launch_bounds__(192) void node_ln192(
    float* __restrict__ acc, const float* __restrict__ S, const float* __restrict__ bias,
    const float* __restrict__ g, const float* __restrict__ be)
{
    __shared__ float red[256];
    int n = blockIdx.x, t = threadIdx.x;
    float s = S[n * 3 + (t >> 6)];
    float v = acc[(size_t)n * HC + t] / (s + 1e-16f) + bias[t];
    v = lrelu(v, 0.01f);
    red[t] = v; if (t < 64) red[192 + t] = 0.f;
    __syncthreads();
    for (int off = 128; off > 0; off >>= 1) { if (t < off) red[t] += red[t + off]; __syncthreads(); }
    float mu = red[0] * (1.f / 192.f);
    __syncthreads();
    float dv = v - mu;
    red[t] = dv * dv; if (t < 64) red[192 + t] = 0.f;
    __syncthreads();
    for (int off = 128; off > 0; off >>= 1) { if (t < off) red[t] += red[t + off]; __syncthreads(); }
    float var = red[0] * (1.f / 192.f);
    acc[(size_t)n * HC + t] = dv * rsqrtf(var + 1e-5f) * g[t] + be[t];
}

// -------- layer2 node pass: normalize, head-mean, +b2, leaky, LayerNorm(64) -----
__global__ __launch_bounds__(192) void node_ln64(
    const float* __restrict__ acc, const float* __restrict__ S, const float* __restrict__ b2,
    const float* __restrict__ g, const float* __restrict__ be, float* __restrict__ out)
{
    __shared__ float sh[192];
    __shared__ float red[64];
    int n = blockIdx.x, t = threadIdx.x;
    float s = S[n * 3 + (t >> 6)];
    sh[t] = acc[(size_t)n * HC + t] / (s + 1e-16f);
    __syncthreads();
    float v = 0.f;
    if (t < 64) {
        v = (sh[t] + sh[64 + t] + sh[128 + t]) * (1.f / 3.f) + b2[t];
        v = lrelu(v, 0.01f);
        red[t] = v;
    }
    __syncthreads();
    for (int off = 32; off > 0; off >>= 1) { if (t < off) red[t] += red[t + off]; __syncthreads(); }
    float mu = red[0] * (1.f / 64.f);
    __syncthreads();
    if (t < 64) { float dv = v - mu; red[t] = dv * dv; }
    __syncthreads();
    for (int off = 32; off > 0; off >>= 1) { if (t < off) red[t] += red[t + off]; __syncthreads(); }
    float var = red[0] * (1.f / 64.f);
    if (t < 64) out[(size_t)n * 64 + t] = (v - mu) * rsqrtf(var + 1e-5f) * g[t] + be[t];
}

extern "C" void kernel_launch(void* const* d_in, const int* in_sizes, int n_in,
                              void* d_out, int out_size, void* d_ws, size_t ws_size,
                              hipStream_t stream) {
    const float* x_s  = (const float*)d_in[0];
    const float* x_t  = (const float*)d_in[1];
    const int*   ei   = (const int*)  d_in[2];   // [2,E]: row0=src, row1=dst
    const float* ea   = (const float*)d_in[3];
    const float* W1   = (const float*)d_in[4];
    const float* a_s1 = (const float*)d_in[5];
    const float* a_d1 = (const float*)d_in[6];
    const float* We1  = (const float*)d_in[7];
    const float* a_e1 = (const float*)d_in[8];
    const float* b1   = (const float*)d_in[9];
    const float* Ws2  = (const float*)d_in[10];
    const float* Wd2  = (const float*)d_in[11];
    const float* a_s2 = (const float*)d_in[12];
    const float* a_d2 = (const float*)d_in[13];
    const float* We2  = (const float*)d_in[14];
    const float* a_e2 = (const float*)d_in[15];
    const float* b2   = (const float*)d_in[16];
    const float* g0   = (const float*)d_in[17];
    const float* be0  = (const float*)d_in[18];
    const float* g1   = (const float*)d_in[19];
    const float* be1  = (const float*)d_in[20];
    float* out = (float*)d_out;

    const int* eSrc = ei;
    const int* eDst = ei + E_EDGES;

    // d_out layout: x_s | at_s | x_t | at_t
    const size_t XS_OFF = 0;
    const size_t ATS_OFF = (size_t)N_NODES * 64;                 // 1,280,000
    const size_t XT_OFF  = ATS_OFF + (size_t)E_EDGES * 3;        // 2,780,000
    const size_t ATT_OFF = XT_OFF + (size_t)N_NODES * 64;        // 4,060,000

    // workspace layout (floats)
    const size_t NB = (size_t)N_NODES * HC;                      // 3,840,000
    float* B0 = (float*)d_ws;            // Hs1 -> Hs_t2
    float* B1 = B0 + NB;                 // Ht1 -> Hs_s2
    float* B2 = B1 + NB;                 // accT1 -> xln_t -> accT2
    float* B3 = B2 + NB;                 // accS1 -> xln_s -> accS2
    float* sc = B3 + NB;
    const int NH = N_NODES * 3;          // 60,000
    float* asrcT = sc;                   // scalar arrays (reused across layers)
    float* adstS = sc + NH;
    float* asrcS = sc + 2 * NH;
    float* adstT = sc + 3 * NH;
    float* S_t   = sc + 4 * NH;
    float* S_s   = sc + 5 * NH;
    unsigned* M_t = (unsigned*)(sc + 6 * NH);
    unsigned* M_s = (unsigned*)(sc + 7 * NH);
    float* U2 = sc + 8 * NH;             // 576
    float* ce = U2 + 576;                // 6: [0..2]=layer1, [3..5]=layer2

    const int GEMM_BLOCKS = (N_NODES / 16) * (HC / 16) / 4;      // 3750 (4 waves/block)
    const int NODE_GRID = (N_NODES + 255) / 256;
    const int EDGE_GRID = (E_EDGES + 255) / 256;

    // ---------------- Layer 1 ----------------
    gemm_wmma_f32<<<GEMM_BLOCKS, 128, 0, stream>>>(x_s, W1, B0, N_NODES, HC, D_IN);
    gemm_wmma_f32<<<GEMM_BLOCKS, 128, 0, stream>>>(x_t, W1, B1, N_NODES, HC, D_IN);
    prep_consts<<<1, 192, 0, stream>>>(Wd2, a_d2, We1, a_e1, We2, a_e2, U2, ce);
    rowdot_head<<<NODE_GRID, 256, 0, stream>>>(B0, a_s1, asrcT, N_NODES); // asrc dir t
    rowdot_head<<<NODE_GRID, 256, 0, stream>>>(B0, a_d1, adstS, N_NODES); // adst dir s
    rowdot_head<<<NODE_GRID, 256, 0, stream>>>(B1, a_s1, asrcS, N_NODES); // asrc dir s
    rowdot_head<<<NODE_GRID, 256, 0, stream>>>(B1, a_d1, adstT, N_NODES); // adst dir t
    init_bufs<<<3750, 256, 0, stream>>>(B2, B3, S_t, S_s, M_t, M_s, (int)NB, NH);
    edge_max<<<EDGE_GRID, 256, 0, stream>>>(eSrc, eDst, asrcT, adstT, ea, ce, M_t, E_EDGES);
    edge_max<<<EDGE_GRID, 256, 0, stream>>>(eDst, eSrc, asrcS, adstS, ea, ce, M_s, E_EDGES);
    edge_acc<<<E_EDGES, 192, 0, stream>>>(eSrc, eDst, asrcT, adstT, ea, ce, M_t, S_t, B0, B2);
    edge_acc<<<E_EDGES, 192, 0, stream>>>(eDst, eSrc, asrcS, adstS, ea, ce, M_s, S_s, B1, B3);
    node_ln192<<<N_NODES, 192, 0, stream>>>(B2, S_t, b1, g0, be0);  // -> xln_t
    node_ln192<<<N_NODES, 192, 0, stream>>>(B3, S_s, b1, g0, be0);  // -> xln_s

    // ---------------- Layer 2 ----------------
    gemm_wmma_f32<<<GEMM_BLOCKS, 128, 0, stream>>>(B3, Ws2, B0, N_NODES, HC, HC); // hs for dir t
    gemm_wmma_f32<<<GEMM_BLOCKS, 128, 0, stream>>>(B2, Ws2, B1, N_NODES, HC, HC); // hs for dir s
    rowdot_head<<<NODE_GRID, 256, 0, stream>>>(B0, a_s2, asrcT, N_NODES);
    rowdot_head<<<NODE_GRID, 256, 0, stream>>>(B1, a_s2, asrcS, N_NODES);
    matvec3<<<NODE_GRID, 256, 0, stream>>>(B2, U2, adstT, N_NODES);  // xln_t @ (Wd2.a_d2)
    matvec3<<<NODE_GRID, 256, 0, stream>>>(B3, U2, adstS, N_NODES);  // xln_s @ (Wd2.a_d2)
    init_bufs<<<3750, 256, 0, stream>>>(B2, B3, S_t, S_s, M_t, M_s, (int)NB, NH);
    edge_max<<<EDGE_GRID, 256, 0, stream>>>(eSrc, eDst, asrcT, adstT, ea, ce + 3, M_t, E_EDGES);
    edge_max<<<EDGE_GRID, 256, 0, stream>>>(eDst, eSrc, asrcS, adstS, ea, ce + 3, M_s, E_EDGES);
    edge_acc<<<E_EDGES, 192, 0, stream>>>(eSrc, eDst, asrcT, adstT, ea, ce + 3, M_t, S_t, B0, B2);
    edge_acc<<<E_EDGES, 192, 0, stream>>>(eDst, eSrc, asrcS, adstS, ea, ce + 3, M_s, S_s, B1, B3);
    edge_alpha<<<EDGE_GRID, 256, 0, stream>>>(eSrc, eDst, asrcT, adstT, ea, ce + 3, M_t, S_t, out + ATT_OFF, E_EDGES);
    edge_alpha<<<EDGE_GRID, 256, 0, stream>>>(eDst, eSrc, asrcS, adstS, ea, ce + 3, M_s, S_s, out + ATS_OFF, E_EDGES);
    node_ln64<<<N_NODES, 192, 0, stream>>>(B2, S_t, b2, g1, be1, out + XT_OFF);
    node_ln64<<<N_NODES, 192, 0, stream>>>(B3, S_s, b2, g1, be1, out + XS_OFF);
}
